// MultiHeadWindowAttention_35132832481953
// MI455X (gfx1250) — matrix operand order repaired and checked by reading, not compiled, not run on hardware
//
#include <hip/hip_runtime.h>

// ---------------------------------------------------------------------------
// MultiHeadWindowAttention for MI455X (gfx1250): bf16 WMMA pipeline
//   B=4 L=4096 DM=1024 H=16 DK=DV=64 WIN=256  -> TOK=16384 tokens
// ---------------------------------------------------------------------------

typedef __bf16 bf16_t;
typedef __attribute__((ext_vector_type(8)))  __bf16 v8bf;
typedef __attribute__((ext_vector_type(16))) __bf16 v16bf;
typedef __attribute__((ext_vector_type(8)))  float  v8f;

#define TOK    16384
#define DMODEL 1024
#define NH     16
#define HD     64
#define WINSZ  256
#define NWIN   16
#define LNEPS  1e-6f

// ---------------------------------------------------------------------------
// WMMA helpers (CDNA5 V_WMMA_F32_16X16X32_BF16, wave32)
// ---------------------------------------------------------------------------
__device__ __forceinline__ v16bf cat8(v8bf lo, v8bf hi) {
  return __builtin_shufflevector(lo, hi, 0,1,2,3,4,5,6,7,8,9,10,11,12,13,14,15);
}

// A operand: 16x32 bf16 tile from a row-major matrix (row stride ld elements).
// ISA: lanes 0-15 hold M=lane, K={0..7,16..23}; lanes 16-31 K={8..15,24..31}.
__device__ __forceinline__ v16bf load_a_tile(const bf16_t* base, int ld, int lane) {
  const int m  = lane & 15;
  const int e0 = (lane < 16) ? 0 : 8;
  const bf16_t* p = base + (size_t)m * ld + e0;
  v8bf lo = *(const v8bf*)(p);
  v8bf hi = *(const v8bf*)(p + 16);
  return cat8(lo, hi);
}

// B operand: 32x16 bf16 tile from an N-major buffer Bt[n][k] (row stride ld).
// ISA: lane n<16 holds col N=n, K={0..7,16..23}; lanes 16-31 K={8..15,24..31}.
__device__ __forceinline__ v16bf load_b_tile(const bf16_t* base, int ld, int lane) {
  const int n  = lane & 15;
  const int k0 = (lane < 16) ? 0 : 8;
  const bf16_t* p = base + (size_t)n * ld + k0;
  v8bf lo = *(const v8bf*)(p);
  v8bf hi = *(const v8bf*)(p + 16);
  return cat8(lo, hi);
}

__device__ __forceinline__ v8f wmma_bf16(v16bf a, v16bf b, v8f c) {
  return __builtin_amdgcn_wmma_f32_16x16x32_bf16(
      /*neg_a=*/false, a, /*neg_b=*/false, b,
      /*c_mod=*/(short)0, c, /*reuse_a=*/false, /*reuse_b=*/false);
}

// ---------------------------------------------------------------------------
// Pre-pass kernels
// ---------------------------------------------------------------------------

// LayerNorm(query) -> bf16, one 256-thread block per token row (1024 cols).
__global__ __launch_bounds__(256)
void ln_kernel(const float* __restrict__ x, const float* __restrict__ gamma,
               const float* __restrict__ beta, bf16_t* __restrict__ out) {
  __shared__ float ssum[256];
  __shared__ float ssq[256];
  const int row = blockIdx.x;
  const int tid = threadIdx.x;
  const float* xr = x + (size_t)row * DMODEL;
  float v[4], s = 0.f, s2 = 0.f;
#pragma unroll
  for (int i = 0; i < 4; ++i) {
    float t = xr[tid + i * 256];
    v[i] = t; s += t; s2 += t * t;
  }
  ssum[tid] = s; ssq[tid] = s2;
  __syncthreads();
  for (int st = 128; st > 0; st >>= 1) {
    if (tid < st) { ssum[tid] += ssum[tid + st]; ssq[tid] += ssq[tid + st]; }
    __syncthreads();
  }
  const float mu  = ssum[0] * (1.f / DMODEL);
  const float var = ssq[0] * (1.f / DMODEL) - mu * mu;
  const float rs  = rsqrtf(var + LNEPS);
  bf16_t* orow = out + (size_t)row * DMODEL;
#pragma unroll
  for (int i = 0; i < 4; ++i) {
    int c = tid + i * 256;
    orow[c] = (bf16_t)((v[i] - mu) * rs * gamma[c] + beta[c]);
  }
}

// float -> bf16 elementwise
__global__ __launch_bounds__(256)
void cvt_kernel(const float* __restrict__ in, bf16_t* __restrict__ out, int n) {
  int i = blockIdx.x * 256 + threadIdx.x;
  if (i < n) out[i] = (bf16_t)in[i];
}

// W[K][N] f32 (row-major) -> Wt[N][K] bf16 (N-major, for B operands)
__global__ __launch_bounds__(256)
void wtrans_kernel(const float* __restrict__ W, bf16_t* __restrict__ Wt, int K, int N) {
  int idx = blockIdx.x * 256 + threadIdx.x;
  if (idx >= K * N) return;
  int n = idx / K, k = idx % K;
  Wt[idx] = (bf16_t)W[(size_t)k * N + n];
}

// ---------------------------------------------------------------------------
// Tiled WMMA GEMM: C[M][N] = A[M][K] * B (B given N-major as Bt[N][K])
// Block = 256 threads = 8 waves arranged 2(M) x 4(N); block tile 64M x 256N;
// each wave owns a 32x64 tile (2 m-tiles x 4 n-tiles of 16x16).
// ---------------------------------------------------------------------------
__device__ __forceinline__ void gemm_core(const bf16_t* __restrict__ A,
                                          const bf16_t* __restrict__ Bt,
                                          int K, int mblk, int nblk, int lane,
                                          v8f acc[2][4]) {
  for (int kk = 0; kk < K; kk += 32) {
    // Unconditional speculative prefetch of the next K-chunk
    // (global_prefetch_b8; OOB translation failures are silently dropped).
    __builtin_prefetch(A  + (size_t)(mblk + lane) * K      + kk + 32, 0, 3);
    __builtin_prefetch(Bt + (size_t)(nblk + lane) * K      + kk + 32, 0, 3);
    __builtin_prefetch(Bt + (size_t)(nblk + 32 + lane) * K + kk + 32, 0, 3);
    v16bf a[2], b[4];
#pragma unroll
    for (int mt = 0; mt < 2; ++mt)
      a[mt] = load_a_tile(A + (size_t)(mblk + mt * 16) * K + kk, K, lane);
#pragma unroll
    for (int nt = 0; nt < 4; ++nt)
      b[nt] = load_b_tile(Bt + (size_t)(nblk + nt * 16) * K + kk, K, lane);
#pragma unroll
    for (int mt = 0; mt < 2; ++mt)
#pragma unroll
      for (int nt = 0; nt < 4; ++nt)
        acc[mt][nt] = wmma_bf16(a[mt], b[nt], acc[mt][nt]);
  }
}

__global__ __launch_bounds__(256)
void gemm_bf16_kernel(const bf16_t* __restrict__ A, const bf16_t* __restrict__ Bt,
                      bf16_t* __restrict__ C, int M, int N, int K) {
  const int lane = threadIdx.x & 31;
  const int wave = threadIdx.x >> 5;
  const int mblk = blockIdx.x * 64  + (wave & 1) * 32;
  const int nblk = blockIdx.y * 256 + (wave >> 1) * 64;
  v8f acc[2][4] = {};
  gemm_core(A, Bt, K, mblk, nblk, lane, acc);
  const int nlo = lane & 15;
  const int rb  = (lane < 16) ? 0 : 8;
#pragma unroll
  for (int mt = 0; mt < 2; ++mt)
#pragma unroll
    for (int nt = 0; nt < 4; ++nt)
#pragma unroll
      for (int j = 0; j < 8; ++j) {
        int row = mblk + mt * 16 + rb + j;
        int col = nblk + nt * 16 + nlo;
        C[(size_t)row * N + col] = (bf16_t)acc[mt][nt][j];
      }
}

// V-projection GEMM that stores V TRANSPOSED in global:
//   VtG[((b*NH + h)*HD + d) * L + pos],  col = h*HD + d, row = b*L + pos
// so attention's ctx GEMM can read its B operands (V^T) as contiguous runs.
__global__ __launch_bounds__(256)
void gemm_bf16_vt_kernel(const bf16_t* __restrict__ A, const bf16_t* __restrict__ Bt,
                         bf16_t* __restrict__ VtG, int M, int N, int K) {
  const int lane = threadIdx.x & 31;
  const int wave = threadIdx.x >> 5;
  const int mblk = blockIdx.x * 64  + (wave & 1) * 32;
  const int nblk = blockIdx.y * 256 + (wave >> 1) * 64;
  v8f acc[2][4] = {};
  gemm_core(A, Bt, K, mblk, nblk, lane, acc);
  const int nlo = lane & 15;
  const int rb  = (lane < 16) ? 0 : 8;
#pragma unroll
  for (int mt = 0; mt < 2; ++mt)
#pragma unroll
    for (int nt = 0; nt < 4; ++nt)
#pragma unroll
      for (int j = 0; j < 8; ++j) {
        int row = mblk + mt * 16 + rb + j;     // token index
        int col = nblk + nt * 16 + nlo;        // h*64 + d
        int bb  = row >> 12, pos = row & 4095;
        int hh  = col >> 6,  dd  = col & 63;
        VtG[((size_t)(bb * NH + hh) * HD + dd) * 4096 + pos] = (bf16_t)acc[mt][nt][j];
      }
}

// Final projection: f32 output with residual add.
__global__ __launch_bounds__(256)
void gemm_out_kernel(const bf16_t* __restrict__ A, const bf16_t* __restrict__ Bt,
                     const float* __restrict__ res, float* __restrict__ C,
                     int M, int N, int K) {
  const int lane = threadIdx.x & 31;
  const int wave = threadIdx.x >> 5;
  const int mblk = blockIdx.x * 64  + (wave & 1) * 32;
  const int nblk = blockIdx.y * 256 + (wave >> 1) * 64;
  v8f acc[2][4] = {};
  gemm_core(A, Bt, K, mblk, nblk, lane, acc);
  const int nlo = lane & 15;
  const int rb  = (lane < 16) ? 0 : 8;
#pragma unroll
  for (int mt = 0; mt < 2; ++mt)
#pragma unroll
    for (int nt = 0; nt < 4; ++nt)
#pragma unroll
      for (int j = 0; j < 8; ++j) {
        size_t idx = (size_t)(mblk + mt * 16 + rb + j) * N + nblk + nt * 16 + nlo;
        C[idx] = acc[mt][nt][j] + res[idx];
      }
}

// ---------------------------------------------------------------------------
// Windowed attention: one block per (b, h, window). 256 threads = 8 waves.
// LDS: scores/probs bf16 [256][256] (128KB). V^T comes straight from global.
// ---------------------------------------------------------------------------
__global__ __launch_bounds__(256)
void attn_kernel(const bf16_t* __restrict__ Qb, const bf16_t* __restrict__ Kb,
                 const bf16_t* __restrict__ VtG, float* __restrict__ attn_out,
                 bf16_t* __restrict__ ctx) {
  extern __shared__ char smem_raw[];
  bf16_t* sc = (bf16_t*)smem_raw;                 // [256][256] scores -> probs

  const int bid  = blockIdx.x;                    // ((b*16 + h)*16 + w)
  const int w    = bid & 15;
  const int h    = (bid >> 4) & 15;
  const int b    = bid >> 8;
  const int bh   = bid >> 4;                      // b*16 + h
  const int tid  = threadIdx.x;
  const int lane = tid & 31;
  const int wv   = tid >> 5;
  const int rowbase = b * 4096 + w * WINSZ;       // token index of window row 0
  const int colh    = h * HD;

  // Phase A: scores = (Q K^T) / 8 -> bf16 in LDS. Wave wv owns rows wv*32..+31.
  // Q A-operands for both m-tiles are hoisted; each K-tile pair is loaded once.
  const int mbase = wv * 32;
  {
    v16bf aq[2][2];
#pragma unroll
    for (int mt = 0; mt < 2; ++mt) {
      const bf16_t* arow = Qb + (size_t)(rowbase + mbase + mt * 16) * DMODEL + colh;
      aq[mt][0] = load_a_tile(arow,      DMODEL, lane);
      aq[mt][1] = load_a_tile(arow + 32, DMODEL, lane);
    }
    const int nlo = lane & 15;
    const int rb  = (lane < 16) ? 0 : 8;
    for (int nt = 0; nt < 16; ++nt) {
      const bf16_t* brow = Kb + (size_t)(rowbase + nt * 16) * DMODEL + colh;
      v16bf b0 = load_b_tile(brow,      DMODEL, lane);
      v16bf b1 = load_b_tile(brow + 32, DMODEL, lane);
#pragma unroll
      for (int mt = 0; mt < 2; ++mt) {
        v8f acc = {};
        acc = wmma_bf16(aq[mt][0], b0, acc);
        acc = wmma_bf16(aq[mt][1], b1, acc);
#pragma unroll
        for (int j = 0; j < 8; ++j)
          sc[(mbase + mt * 16 + rb + j) * WINSZ + nt * 16 + nlo] =
              (bf16_t)(acc[j] * 0.125f);
      }
    }
  }
  __syncthreads();

  // Phase B: row softmax, thread t owns row t; all LDS traffic is b128.
  {
    bf16_t* row = sc + tid * WINSZ;
    float mx = -3.0e38f;
    for (int c = 0; c < WINSZ / 8; ++c) {
      v8bf x = *(const v8bf*)(row + c * 8);
#pragma unroll
      for (int j = 0; j < 8; ++j) mx = fmaxf(mx, (float)x[j]);
    }
    float sum = 0.f;
    for (int c = 0; c < WINSZ / 8; ++c) {          // exp pass (unnormalized)
      v8bf x = *(const v8bf*)(row + c * 8);
      v8bf e;
#pragma unroll
      for (int j = 0; j < 8; ++j) {
        float p = __expf((float)x[j] - mx);
        sum += p;
        e[j] = (bf16_t)p;
      }
      *(v8bf*)(row + c * 8) = e;
    }
    const float inv = 1.f / sum;
    for (int c = 0; c < WINSZ / 8; ++c) {          // normalize in place
      v8bf x = *(const v8bf*)(row + c * 8);
      v8bf p;
#pragma unroll
      for (int j = 0; j < 8; ++j) p[j] = (bf16_t)((float)x[j] * inv);
      *(v8bf*)(row + c * 8) = p;
    }
  }
  __syncthreads();

  // Coalesced f32 attn output: consecutive lanes write consecutive 8-col
  // chunks of the same row (b128 global stores).
  {
    float* abase = attn_out + (size_t)bid * WINSZ * WINSZ;
    for (int i = 0; i < 32; ++i) {
      int idx = tid + 256 * i;                     // 8192 (row, chunk) tasks
      int r   = idx >> 5;
      int c8  = idx & 31;
      v8bf x = *(const v8bf*)(sc + r * WINSZ + c8 * 8);
      float4 lo, hi;
      lo.x = (float)x[0]; lo.y = (float)x[1]; lo.z = (float)x[2]; lo.w = (float)x[3];
      hi.x = (float)x[4]; hi.y = (float)x[5]; hi.z = (float)x[6]; hi.w = (float)x[7];
      float4* dst = (float4*)(abase + (size_t)r * WINSZ + c8 * 8);
      dst[0] = lo; dst[1] = hi;
    }
  }

  // Phase C: ctx = P @ V. A = probs (LDS), B = V^T straight from global VtG.
  v8f acc[2][4] = {};
  for (int kt = 0; kt < 8; ++kt) {                 // 8 chunks of 32 keys
    v16bf a[2], bb[4];
#pragma unroll
    for (int mt = 0; mt < 2; ++mt)
      a[mt] = load_a_tile(sc + (mbase + mt * 16) * WINSZ + kt * 32, WINSZ, lane);
#pragma unroll
    for (int nt = 0; nt < 4; ++nt)
      bb[nt] = load_b_tile(
          VtG + ((size_t)bh * HD + nt * 16) * 4096 + w * WINSZ + kt * 32,
          4096, lane);
#pragma unroll
    for (int mt = 0; mt < 2; ++mt)
#pragma unroll
      for (int nt = 0; nt < 4; ++nt)
        acc[mt][nt] = wmma_bf16(a[mt], bb[nt], acc[mt][nt]);
  }
  const int nlo = lane & 15;
  const int rb  = (lane < 16) ? 0 : 8;
#pragma unroll
  for (int mt = 0; mt < 2; ++mt)
#pragma unroll
    for (int nt = 0; nt < 4; ++nt)
#pragma unroll
      for (int j = 0; j < 8; ++j) {
        int r = mbase + mt * 16 + rb + j;
        int d = nt * 16 + nlo;
        ctx[(size_t)(rowbase + r) * DMODEL + colh + d] = (bf16_t)acc[mt][nt][j];
      }
}

// ---------------------------------------------------------------------------
// Host-side launcher
// ---------------------------------------------------------------------------
extern "C" void kernel_launch(void* const* d_in, const int* in_sizes, int n_in,
                              void* d_out, int out_size, void* d_ws, size_t ws_size,
                              hipStream_t stream) {
  const float* q   = (const float*)d_in[0];
  const float* key = (const float*)d_in[1];
  const float* val = (const float*)d_in[2];
  const float* Wq  = (const float*)d_in[3];
  const float* Wk  = (const float*)d_in[4];
  const float* Wv  = (const float*)d_in[5];
  const float* Wo  = (const float*)d_in[6];
  const float* g   = (const float*)d_in[7];
  const float* be  = (const float*)d_in[8];

  float* out  = (float*)d_out;
  float* attn = out + (size_t)TOK * DMODEL;       // [B,H,nW,W,W] f32

  const size_t SZ = (size_t)TOK * DMODEL;
  bf16_t* ws  = (bf16_t*)d_ws;
  bf16_t* xq  = ws;               // LN(query) bf16; reused as ctx after Q GEMM
  bf16_t* kbf = xq  + SZ;
  bf16_t* vbf = kbf + SZ;
  bf16_t* Qb  = vbf + SZ;
  bf16_t* Kb  = Qb  + SZ;
  bf16_t* VtG = Kb  + SZ;         // V^T: [B][H][64][L]
  bf16_t* Wqt = VtG + SZ;
  bf16_t* Wkt = Wqt + (size_t)DMODEL * DMODEL;
  bf16_t* Wvt = Wkt + (size_t)DMODEL * DMODEL;
  bf16_t* Wot = Wvt + (size_t)DMODEL * DMODEL;
  bf16_t* ctx = xq;               // safe: xq consumed by Q GEMM before attn runs

  const int nElem   = TOK * DMODEL;
  const int cvtGrid = (nElem + 255) / 256;
  const int wGrid   = (DMODEL * DMODEL + 255) / 256;

  // Pre-passes
  ln_kernel<<<TOK, 256, 0, stream>>>(q, g, be, xq);
  cvt_kernel<<<cvtGrid, 256, 0, stream>>>(key, kbf, nElem);
  cvt_kernel<<<cvtGrid, 256, 0, stream>>>(val, vbf, nElem);
  wtrans_kernel<<<wGrid, 256, 0, stream>>>(Wq, Wqt, DMODEL, DMODEL);
  wtrans_kernel<<<wGrid, 256, 0, stream>>>(Wk, Wkt, DMODEL, DMODEL);
  wtrans_kernel<<<wGrid, 256, 0, stream>>>(Wv, Wvt, DMODEL, DMODEL);
  wtrans_kernel<<<wGrid, 256, 0, stream>>>(Wo, Wot, DMODEL, DMODEL);

  // Projections (WMMA); V goes out transposed.
  dim3 ggrid(TOK / 64, DMODEL / 256);
  gemm_bf16_kernel   <<<ggrid, 256, 0, stream>>>(xq,  Wqt, Qb,  TOK, DMODEL, DMODEL);
  gemm_bf16_kernel   <<<ggrid, 256, 0, stream>>>(kbf, Wkt, Kb,  TOK, DMODEL, DMODEL);
  gemm_bf16_vt_kernel<<<ggrid, 256, 0, stream>>>(vbf, Wvt, VtG, TOK, DMODEL, DMODEL);

  // Windowed attention (WMMA + softmax), 1024 blocks = B*H*nW, 128KB LDS
  const size_t lds = (size_t)WINSZ * WINSZ * 2;
  attn_kernel<<<4 * NH * NWIN, 256, lds, stream>>>(Qb, Kb, VtG, attn, ctx);

  // Output projection + residual (WMMA, f32 out)
  gemm_out_kernel<<<ggrid, 256, 0, stream>>>(ctx, Wot, q, out, TOK, DMODEL, DMODEL);
}
